// SE3TransformerModel_84344567759033
// MI455X (gfx1250) — matrix operand order repaired and checked by reading, compiled, not run on hardware
//
#include <hip/hip_runtime.h>
#include <hip/hip_bf16.h>
#include <math.h>

// ---------------- model constants ----------------
#define B_  8
#define N_  256
#define C_  128          // feature dim == F
#define H_  8
#define DH_ 16
#define T1_ 2
#define MODES_ 6
#define TOUT_ 80
#define EPSV 1e-6f
#define SCALE_QK 0.17677669529663688f   // 1/sqrt(2*DH)

typedef __attribute__((ext_vector_type(16))) _Float16 v16h;
typedef __attribute__((ext_vector_type(8)))  _Float16 v8h;
typedef __attribute__((ext_vector_type(4)))  _Float16 v4h;
typedef __attribute__((ext_vector_type(8)))  float    v8f;

// ---------------- WMMA helpers (CDNA5 16x16x32 f16 -> f32) ----------------
__device__ __forceinline__ v8f wmma16(v16h a, v16h b, v8f c) {
  // 8 args: (neg_a, A, neg_b, B, c_mod, C, reuse_a, reuse_b)
  return __builtin_amdgcn_wmma_f32_16x16x32_f16(false, a, false, b, (short)0, c,
                                                false, false);
}

// A fragment: lane<16 holds row M=lane, K{k0+0..7, k0+16..23}; lane>=16: K{+8..15, +24..31}
__device__ __forceinline__ v16h frag_a(const _Float16* row, int k0, int half) {
  v8h lo = *(const v8h*)(row + k0 + 8 * half);
  v8h hi = *(const v8h*)(row + k0 + 16 + 8 * half);
  v16h r;
#pragma unroll
  for (int i = 0; i < 8; ++i) { r[i] = lo[i]; r[i + 8] = hi[i]; }
  return r;
}

// B fragment: lane<16 holds col N=lane, K{k0+0..7, k0+16..23}; lane>=16: K{+8..15, +24..31}
__device__ __forceinline__ v16h frag_b(const _Float16* col, int k0, int half) {
  const _Float16* p = col + k0 + 8 * half;
  v8h lo = *(const v8h*)(p);
  v8h hi = *(const v8h*)(p + 16);
  v16h r;
#pragma unroll
  for (int i = 0; i < 8; ++i) { r[i] = lo[i]; r[i + 8] = hi[i]; }
  return r;
}

// ---------------- preprocessing ----------------
// f0[b,n,c] ; f1 stored [b,n,x,c] (x = xyz)
__global__ void k_preprocess(const int* __restrict__ valid,
                             const int* __restrict__ t0,
                             const float* __restrict__ type1,
                             const float* __restrict__ emb,
                             const float* __restrict__ w_t1,
                             float* __restrict__ f0, float* __restrict__ f1) {
  int bn = blockIdx.x;            // B*N
  int c  = threadIdx.x;           // C
  int v  = valid[bn];
  int tok = t0[bn] * v;
  f0[bn * C_ + c] = emb[tok * C_ + c] * (float)v;
  const float* t1p = type1 + bn * T1_ * 3;
#pragma unroll
  for (int x = 0; x < 3; ++x) {
    float s = 0.f;
#pragma unroll
    for (int k = 0; k < T1_; ++k) s += t1p[k * 3 + x] * w_t1[k * C_ + c];
    f1[(bn * 3 + x) * C_ + c] = s * (float)v;
  }
}

__global__ void k_geometry(const float* __restrict__ coords,
                           float* __restrict__ dist, float* __restrict__ rhat) {
  int idx = blockIdx.x * blockDim.x + threadIdx.x;  // b*N*N + i*N + j
  int j = idx & (N_ - 1);
  int i = (idx >> 8) & (N_ - 1);
  int b = idx >> 16;
  float dx = coords[(b * N_ + i) * 3 + 0] - coords[(b * N_ + j) * 3 + 0];
  float dy = coords[(b * N_ + i) * 3 + 1] - coords[(b * N_ + j) * 3 + 1];
  float dz = coords[(b * N_ + i) * 3 + 2] - coords[(b * N_ + j) * 3 + 2];
  float d = sqrtf(dx * dx + dy * dy + dz * dz + EPSV);
  dist[idx] = d;
  float inv = 1.0f / d;
  rhat[idx * 3 + 0] = dx * inv;
  rhat[idx * 3 + 1] = dy * inv;
  rhat[idx * 3 + 2] = dz * inv;
}

// ---------------- equivariant norms (f32 in, f16 out for WMMA) ----------------
__global__ void k_eqnorm0(const float* __restrict__ f0,
                          const float* __restrict__ g,
                          _Float16* __restrict__ Xh) {
  __shared__ float sred[C_];
  int bn = blockIdx.x, c = threadIdx.x;
  float x = f0[bn * C_ + c];
  sred[c] = x * x;
  __syncthreads();
  for (int s = C_ / 2; s > 0; s >>= 1) {
    if (c < s) sred[c] += sred[c + s];
    __syncthreads();
  }
  float rms = sqrtf(sred[0] * (1.0f / C_) + EPSV);
  Xh[bn * C_ + c] = (_Float16)(x / rms * g[c]);
}

__global__ void k_eqnorm1(const float* __restrict__ f1,
                          const float* __restrict__ g,
                          _Float16* __restrict__ Xh) {
  __shared__ float sred[C_];
  int bn = blockIdx.x, c = threadIdx.x;
  float x0 = f1[(bn * 3 + 0) * C_ + c];
  float x1 = f1[(bn * 3 + 1) * C_ + c];
  float x2 = f1[(bn * 3 + 2) * C_ + c];
  sred[c] = x0 * x0 + x1 * x1 + x2 * x2;
  __syncthreads();
  for (int s = C_ / 2; s > 0; s >>= 1) {
    if (c < s) sred[c] += sred[c + s];
    __syncthreads();
  }
  float rms = sqrtf(sred[0] * (1.0f / C_) + EPSV);
  float sc = g[c] / rms;
  Xh[(bn * 3 + 0) * C_ + c] = (_Float16)(x0 * sc);
  Xh[(bn * 3 + 1) * C_ + c] = (_Float16)(x1 * sc);
  Xh[(bn * 3 + 2) * C_ + c] = (_Float16)(x2 * sc);
}

// ---------------- weight convert: W[K=128, Cout] f32 -> Wt[CoutPad][128] f16, zero pad ----
__global__ void k_convert_w(const float* __restrict__ W, _Float16* __restrict__ Wt,
                            int Cout, int CoutPad) {
  int idx = blockIdx.x * blockDim.x + threadIdx.x;
  if (idx >= 128 * CoutPad) return;
  int n = idx % CoutPad;
  int k = idx / CoutPad;
  float v = (n < Cout) ? W[k * Cout + n] : 0.0f;
  Wt[n * 128 + k] = (_Float16)v;
}

// ---------------- generic GEMM: Y[R,Cout] = X[R,128] @ W[128,Cout] ----------------
// block = 256 threads (8 waves). A-panel of 64 rows staged via async global->LDS.
// wave w: rows (w>>1)*16..+15, cols blockIdx.y*128 + (w&1)*64 .. +63 (4 acc tiles).
// Wt must be zero-padded to a multiple of 64 columns.
__global__ void k_gemm128(const _Float16* __restrict__ X,
                          const _Float16* __restrict__ Wt,
                          float* __restrict__ Y, int R, int Cout) {
  __shared__ _Float16 sX[64 * 128];
  int t = threadIdx.x;
  int tile_r = blockIdx.x;  // 64-row tiles
  const _Float16* gsrc = X + (size_t)tile_r * 64 * 128;
#if defined(__gfx1250__)
  {  // ASYNCcnt path: each lane copies 4 x 16B, no VGPR round-trip
    unsigned lds0 = (unsigned)(size_t)(&sX[0]);
#pragma unroll
    for (int it = 0; it < 4; ++it) {
      int chunk = t + it * 256;                     // 16-byte chunks
      unsigned ldsa = lds0 + chunk * 16;
      const _Float16* g = gsrc + chunk * 8;
      asm volatile("global_load_async_to_lds_b128 %0, %1, off"
                   :: "v"(ldsa), "v"(g)
                   : "memory");
    }
    asm volatile("s_wait_asynccnt 0x0" ::: "memory");
  }
#else
  for (int it = 0; it < 4; ++it) {
    int chunk = t + it * 256;
    *(v8h*)(sX + chunk * 8) = *(const v8h*)(gsrc + chunk * 8);
  }
#endif
  __syncthreads();

  int wave = t >> 5, lane = t & 31;
  int half = lane >> 4, mr = lane & 15;
  int rbase = (wave >> 1) * 16;
  int ncol0 = blockIdx.y * 128 + (wave & 1) * 64;
  if (ncol0 >= Cout) return;  // wave-uniform skip (after barrier)

  const _Float16* arow = sX + (rbase + mr) * 128;
  const _Float16* cp0 = Wt + (size_t)(ncol0 + 0  + mr) * 128;
  const _Float16* cp1 = Wt + (size_t)(ncol0 + 16 + mr) * 128;
  const _Float16* cp2 = Wt + (size_t)(ncol0 + 32 + mr) * 128;
  const _Float16* cp3 = Wt + (size_t)(ncol0 + 48 + mr) * 128;
  v8f acc[4] = {{}, {}, {}, {}};
#pragma unroll
  for (int k0 = 0; k0 < 128; k0 += 32) {
    v16h a = frag_a(arow, k0, half);          // reused across 4 B tiles
    acc[0] = wmma16(a, frag_b(cp0, k0, half), acc[0]);
    acc[1] = wmma16(a, frag_b(cp1, k0, half), acc[1]);
    acc[2] = wmma16(a, frag_b(cp2, k0, half), acc[2]);
    acc[3] = wmma16(a, frag_b(cp3, k0, half), acc[3]);
  }
#pragma unroll
  for (int cg = 0; cg < 4; ++cg) {
    int ocol = ncol0 + cg * 16 + mr;
    if (ocol < Cout) {
#pragma unroll
      for (int r = 0; r < 8; ++r) {
        int orow = tile_r * 64 + rbase + 8 * half + r;  // D: M = 8*half + r
        Y[(size_t)orow * Cout + ocol] = acc[cg][r];
      }
    }
  }
}

// ---------------- pack q/k/v into attention layouts ----------------
// qcat,kcat: [B,H,N,64] f16 (slot0=deg0, slots1..3=deg1 xyz); vT: [B,H,64,N] f16
__global__ void k_pack_qkv(const float* __restrict__ Yq0, const float* __restrict__ Yk0,
                           const float* __restrict__ Yv0, const float* __restrict__ Yq1,
                           const float* __restrict__ Yk1, const float* __restrict__ Yv1,
                           _Float16* __restrict__ qcat, _Float16* __restrict__ kcat,
                           _Float16* __restrict__ vT) {
  int idx = blockIdx.x * blockDim.x + threadIdx.x;  // ((b*H+h)*N+n)*64 + dcat
  int dcat = idx & 63;
  int n = (idx >> 6) & (N_ - 1);
  int h = (idx >> 14) & (H_ - 1);
  int b = idx >> 17;
  int s = dcat >> 4, d = dcat & 15;
  int col = h * DH_ + d;
  int bn = b * N_ + n;
  float q, k, v;
  if (s == 0) {
    q = Yq0[bn * C_ + col]; k = Yk0[bn * C_ + col]; v = Yv0[bn * C_ + col];
  } else {
    int row = bn * 3 + (s - 1);
    q = Yq1[row * C_ + col]; k = Yk1[row * C_ + col]; v = Yv1[row * C_ + col];
  }
  qcat[idx] = (_Float16)(q * SCALE_QK);
  kcat[idx] = (_Float16)k;
  vT[((b * H_ + h) * 64 + dcat) * N_ + n] = (_Float16)v;
}

// ---------------- attention logits: S = Qcat @ Kcat^T  + bias + mask ----------------
// grid: (N/16, N/(16*8), B*H), block 256
__global__ void k_logits(const _Float16* __restrict__ qcat,
                         const _Float16* __restrict__ kcat,
                         const float* __restrict__ dist,
                         const int* __restrict__ valid,
                         const float* __restrict__ wd,
                         float* __restrict__ attn) {
  __shared__ _Float16 sQ[16 * 64];
  int bh = blockIdx.z;
  int b = bh >> 3, h = bh & (H_ - 1);
  int ti = blockIdx.x;
  int t = threadIdx.x;
  {  // stage 16x64 q tile
    int row = t >> 4, chunk = t & 15;
    *(v4h*)(sQ + row * 64 + chunk * 4) =
        *(const v4h*)(qcat + (bh * N_ + ti * 16 + row) * 64 + chunk * 4);
  }
  __syncthreads();
  int wave = t >> 5, lane = t & 31;
  int half = lane >> 4, mr = lane & 15;
  int tj = blockIdx.y * 8 + wave;
  float wdh = wd[h];
  float sp = (wdh > 20.f) ? wdh : log1pf(expf(wdh));  // softplus
  v8f acc = {};
  const _Float16* kcol = kcat + (bh * N_ + tj * 16 + mr) * 64;
#pragma unroll
  for (int k0 = 0; k0 < 64; k0 += 32) {
    v16h a = frag_a(sQ + mr * 64, k0, half);
    v16h bb = frag_b(kcol, k0, half);
    acc = wmma16(a, bb, acc);
  }
  int j = tj * 16 + mr;
  int vj = valid[b * N_ + j];
#pragma unroll
  for (int r = 0; r < 8; ++r) {
    int i = ti * 16 + 8 * half + r;
    float lg = acc[r] - sp * dist[(b * N_ + i) * N_ + j];
    if (vj == 0) lg = -1e9f;
    attn[(bh * N_ + i) * N_ + j] = lg;
  }
}

// ---------------- row softmax over j (N=256) ----------------
__global__ void k_softmax(float* __restrict__ attn) {
  __shared__ float sred[N_];
  int row = blockIdx.x;  // B*H*N rows
  int t = threadIdx.x;
  float x = attn[row * N_ + t];
  sred[t] = x;
  __syncthreads();
  for (int s = N_ / 2; s > 0; s >>= 1) {
    if (t < s) sred[t] = fmaxf(sred[t], sred[t + s]);
    __syncthreads();
  }
  float mx = sred[0];
  __syncthreads();
  float e = expf(x - mx);
  sred[t] = e;
  __syncthreads();
  for (int s = N_ / 2; s > 0; s >>= 1) {
    if (t < s) sred[t] += sred[t + s];
    __syncthreads();
  }
  attn[row * N_ + t] = e / sred[0];
}

// ---------------- attention x values with geometric mixing ----------------
// aggcat[b,h,i,64]: slot0 = A0@v0 + sum_c Ac@v1c ; slot(c+1) = A0@v1c + Ac@v0
// where A0 = attn, Ac = attn * rel_hat_c.  grid: (N/(16*8), B*H), block 256
__global__ void k_av(const float* __restrict__ attn, const float* __restrict__ rhat,
                     const _Float16* __restrict__ vT, float* __restrict__ aggcat) {
  int bh = blockIdx.y;
  int b = bh >> 3;
  int t = threadIdx.x;
  int wave = t >> 5, lane = t & 31;
  int half = lane >> 4, mr = lane & 15;
  int ti = blockIdx.x * 8 + wave;
  int i = ti * 16 + mr;
  const float* arow = attn + (bh * N_ + i) * N_;
  const float* rrow = rhat + (size_t)(b * N_ + i) * N_ * 3;
  const _Float16* vbase = vT + (size_t)bh * 64 * N_;
  v8f acc0 = {}, acc1 = {}, acc2 = {}, acc3 = {};
  for (int jb = 0; jb < N_; jb += 32) {
    if (jb + 32 < N_) {  // prefetch next attention/rhat chunk
      __builtin_prefetch(arow + jb + 32, 0, 0);
      __builtin_prefetch(rrow + (jb + 32) * 3, 0, 0);
    }
    v16h A0, A1, A2, A3;
#pragma unroll
    for (int e = 0; e < 16; ++e) {
      int kk = (e < 8) ? (8 * half + e) : (16 + 8 * half + (e - 8));
      int j = jb + kk;
      float a = arow[j];
      const float* rp = rrow + j * 3;
      A0[e] = (_Float16)a;
      A1[e] = (_Float16)(a * rp[0]);
      A2[e] = (_Float16)(a * rp[1]);
      A3[e] = (_Float16)(a * rp[2]);
    }
    v16h B0 = frag_b(vbase + (0 * 16 + mr) * N_, jb, half);
    v16h B1 = frag_b(vbase + (1 * 16 + mr) * N_, jb, half);
    v16h B2 = frag_b(vbase + (2 * 16 + mr) * N_, jb, half);
    v16h B3 = frag_b(vbase + (3 * 16 + mr) * N_, jb, half);
    acc0 = wmma16(A0, B0, acc0);
    acc0 = wmma16(A1, B1, acc0);
    acc0 = wmma16(A2, B2, acc0);
    acc0 = wmma16(A3, B3, acc0);
    acc1 = wmma16(A0, B1, acc1);
    acc1 = wmma16(A1, B0, acc1);
    acc2 = wmma16(A0, B2, acc2);
    acc2 = wmma16(A2, B0, acc2);
    acc3 = wmma16(A0, B3, acc3);
    acc3 = wmma16(A3, B0, acc3);
  }
#pragma unroll
  for (int r = 0; r < 8; ++r) {
    int orow = ti * 16 + 8 * half + r;
    float* out = aggcat + (bh * N_ + orow) * 64 + mr;
    out[0]  = acc0[r];
    out[16] = acc1[r];
    out[32] = acc2[r];
    out[48] = acc3[r];
  }
}

// ---------------- repack aggregation for output projection ----------------
__global__ void k_pack_agg(const float* __restrict__ aggcat,
                           _Float16* __restrict__ Xh0, _Float16* __restrict__ Xh1) {
  int idx = blockIdx.x * blockDim.x + threadIdx.x;  // same layout as pack_qkv
  int dcat = idx & 63;
  int n = (idx >> 6) & (N_ - 1);
  int h = (idx >> 14) & (H_ - 1);
  int b = idx >> 17;
  int s = dcat >> 4, d = dcat & 15;
  int col = h * DH_ + d;
  int bn = b * N_ + n;
  float v = aggcat[idx];
  if (s == 0) Xh0[bn * C_ + col] = (_Float16)v;
  else        Xh1[(bn * 3 + (s - 1)) * C_ + col] = (_Float16)v;
}

__global__ void k_residual(const int* __restrict__ valid,
                           const float* __restrict__ Yo0, const float* __restrict__ Yo1,
                           float* __restrict__ f0, float* __restrict__ f1) {
  int bn = blockIdx.x, c = threadIdx.x;
  float v = (float)valid[bn];
  f0[bn * C_ + c] += Yo0[bn * C_ + c] * v;
#pragma unroll
  for (int x = 0; x < 3; ++x)
    f1[(bn * 3 + x) * C_ + c] += Yo1[(bn * 3 + x) * C_ + c] * v;
}

// ---------------- final heads ----------------
__global__ void k_logsoftmax(const float* __restrict__ Y0, float* __restrict__ lp) {
  int bn = blockIdx.x;
  if (threadIdx.x != 0) return;
  float m = -1e30f;
  for (int k = 0; k < MODES_; ++k) m = fmaxf(m, Y0[bn * MODES_ + k]);
  float s = 0.f;
  for (int k = 0; k < MODES_; ++k) s += expf(Y0[bn * MODES_ + k] - m);
  float ls = logf(s) + m;
  for (int k = 0; k < MODES_; ++k) lp[bn * MODES_ + k] = Y0[bn * MODES_ + k] - ls;
}

// mode_means[b,n,mode,t,x] = Y1[(bn*3+x)*480 + mode*80 + t]
__global__ void k_scatter_means(const float* __restrict__ Y1, float* __restrict__ means) {
  int idx = blockIdx.x * blockDim.x + threadIdx.x;
  const int total = B_ * N_ * MODES_ * TOUT_ * 3;
  if (idx >= total) return;
  int x = idx % 3;
  int tt = (idx / 3) % TOUT_;
  int mode = (idx / (3 * TOUT_)) % MODES_;
  int bn = idx / (3 * TOUT_ * MODES_);
  means[idx] = Y1[((size_t)bn * 3 + x) * (MODES_ * TOUT_) + mode * TOUT_ + tt];
}

// ====================================================================
extern "C" void kernel_launch(void* const* d_in, const int* in_sizes, int n_in,
                              void* d_out, int out_size, void* d_ws, size_t ws_size,
                              hipStream_t stream) {
  const int BN = B_ * N_;
  const int BN3 = BN * 3;

  // ---- inputs (setup_inputs() insertion order; params flattened recursively) ----
  const int*   valid  = (const int*)d_in[0];
  const int*   t0     = (const int*)d_in[1];
  const float* type1  = (const float*)d_in[2];
  const float* coords = (const float*)d_in[3];
  const float* emb    = (const float*)d_in[4];
  const float* w_t1   = (const float*)d_in[5];
  const float* g0_f   = (const float*)d_in[6];
  const float* g1_f   = (const float*)d_in[7];
  const float* w_out0 = (const float*)d_in[8];
  const float* w_out1 = (const float*)d_in[9];
  // layers: base 10, 11 tensors each: wq0 wk0 wv0 wq1 wk1 wv1 wo0 wo1 wd g0 g1

  // ---- workspace bump allocator ----
  char* ws = (char*)d_ws;
  size_t off = 0;
  auto alloc = [&](size_t bytes) -> void* {
    off = (off + 255) & ~(size_t)255;
    void* p = ws + off;
    off += bytes;
    return p;
  };
  float* f0   = (float*)alloc((size_t)BN * C_ * 4);
  float* f1   = (float*)alloc((size_t)BN3 * C_ * 4);
  float* dist = (float*)alloc((size_t)B_ * N_ * N_ * 4);
  float* rhat = (float*)alloc((size_t)B_ * N_ * N_ * 3 * 4);
  float* attn = (float*)alloc((size_t)B_ * H_ * N_ * N_ * 4);
  float* Yq0  = (float*)alloc((size_t)BN * C_ * 4);
  float* Yk0  = (float*)alloc((size_t)BN * C_ * 4);
  float* Yv0  = (float*)alloc((size_t)BN * C_ * 4);
  float* Yq1  = (float*)alloc((size_t)BN3 * C_ * 4);
  float* Yk1  = (float*)alloc((size_t)BN3 * C_ * 4);
  float* Yv1  = (float*)alloc((size_t)BN3 * C_ * 4);
  float* aggc = (float*)alloc((size_t)B_ * H_ * N_ * 64 * 4);
  float* Yo0  = (float*)alloc((size_t)BN * C_ * 4);
  float* Yo1  = (float*)alloc((size_t)BN3 * C_ * 4);
  _Float16* X0h   = (_Float16*)alloc((size_t)BN * C_ * 2);
  _Float16* X1h   = (_Float16*)alloc((size_t)BN3 * C_ * 2);
  _Float16* Wt    = (_Float16*)alloc((size_t)512 * 128 * 2);  // padded to 64-col multiple
  _Float16* qcat  = (_Float16*)alloc((size_t)B_ * H_ * N_ * 64 * 2);
  _Float16* kcat  = (_Float16*)alloc((size_t)B_ * H_ * N_ * 64 * 2);
  _Float16* vT    = (_Float16*)alloc((size_t)B_ * H_ * 64 * N_ * 2);
  _Float16* aXh0  = (_Float16*)alloc((size_t)BN * C_ * 2);
  _Float16* aXh1  = (_Float16*)alloc((size_t)BN3 * C_ * 2);
  float* Y0f = (float*)alloc((size_t)BN * MODES_ * 4);
  float* Y1f = (float*)alloc((size_t)BN3 * MODES_ * TOUT_ * 4);

  float* out_means = (float*)d_out;
  float* out_lp    = out_means + (size_t)B_ * N_ * MODES_ * TOUT_ * 3;

  auto gemm = [&](const _Float16* X, const float* W, float* Y, int R, int Cout) {
    int cp = (Cout + 63) & ~63;  // pad so every 64-wide wave strip reads zeros
    int tot = 128 * cp;
    k_convert_w<<<(tot + 255) / 256, 256, 0, stream>>>(W, Wt, Cout, cp);
    dim3 g(R / 64, (Cout + 127) / 128);
    k_gemm128<<<g, 256, 0, stream>>>(X, Wt, Y, R, Cout);
  };

  // ---- preprocess + geometry ----
  k_preprocess<<<BN, C_, 0, stream>>>(valid, t0, type1, emb, w_t1, f0, f1);
  k_geometry<<<(B_ * N_ * N_) / 256, 256, 0, stream>>>(coords, dist, rhat);

  // ---- transformer layers ----
  for (int L = 0; L < 4; ++L) {
    const float* const* P = (const float* const*)(d_in + 10 + 11 * L);
    const float *wq0 = P[0], *wk0 = P[1], *wv0 = P[2];
    const float *wq1 = P[3], *wk1 = P[4], *wv1 = P[5];
    const float *wo0 = P[6], *wo1 = P[7], *wd = P[8];
    const float *g0 = P[9], *g1 = P[10];

    k_eqnorm0<<<BN, C_, 0, stream>>>(f0, g0, X0h);
    k_eqnorm1<<<BN, C_, 0, stream>>>(f1, g1, X1h);

    gemm(X0h, wq0, Yq0, BN, C_);
    gemm(X0h, wk0, Yk0, BN, C_);
    gemm(X0h, wv0, Yv0, BN, C_);
    gemm(X1h, wq1, Yq1, BN3, C_);
    gemm(X1h, wk1, Yk1, BN3, C_);
    gemm(X1h, wv1, Yv1, BN3, C_);

    k_pack_qkv<<<(B_ * H_ * N_ * 64) / 256, 256, 0, stream>>>(
        Yq0, Yk0, Yv0, Yq1, Yk1, Yv1, qcat, kcat, vT);

    k_logits<<<dim3(N_ / 16, N_ / (16 * 8), B_ * H_), 256, 0, stream>>>(
        qcat, kcat, dist, valid, wd, attn);
    k_softmax<<<B_ * H_ * N_, N_, 0, stream>>>(attn);
    k_av<<<dim3(N_ / (16 * 8), B_ * H_), 256, 0, stream>>>(attn, rhat, vT, aggc);

    k_pack_agg<<<(B_ * H_ * N_ * 64) / 256, 256, 0, stream>>>(aggc, aXh0, aXh1);
    gemm(aXh0, wo0, Yo0, BN, C_);
    gemm(aXh1, wo1, Yo1, BN3, C_);
    k_residual<<<BN, C_, 0, stream>>>(valid, Yo0, Yo1, f0, f1);
  }

  // ---- output heads ----
  k_eqnorm0<<<BN, C_, 0, stream>>>(f0, g0_f, X0h);
  k_eqnorm1<<<BN, C_, 0, stream>>>(f1, g1_f, X1h);
  gemm(X0h, w_out0, Y0f, BN, MODES_);
  gemm(X1h, w_out1, Y1f, BN3, MODES_ * TOUT_);

  k_logsoftmax<<<BN, 32, 0, stream>>>(Y0f, out_lp);
  const int totm = B_ * N_ * MODES_ * TOUT_ * 3;
  k_scatter_means<<<(totm + 255) / 256, 256, 0, stream>>>(Y1f, out_means);
}
